// Model_12429635355241
// MI455X (gfx1250) — matrix operand order repaired and checked by reading, not compiled, not run on hardware
//
#include <hip/hip_runtime.h>

// Sparse autoencoder forward for MI455X (gfx1250, wave32).
//   K1 enc_gemm : a = (x - b_dec) @ W_enc^T + b_enc, bf16 WMMA, fp32 accum.
//                 a is written into the f region of d_out (used as scratch).
//   K2 topk     : per-row radix-select of top-32 keys, rewrite row in place as
//                 f = scatter(relu(top32)), emit deterministic (idx,val) list.
//   K3 dec      : xhat = b_dec + sum_j val_j * W_dec[:, idx_j]  (sparse decode,
//                 coalesced via W_dec^T in workspace when ws_size permits).

#define FEATS  1024
#define LAT    16384
#define NBATCH 4096
#define TOPK   32

typedef __attribute__((ext_vector_type(16))) __bf16 v16bf;
typedef __attribute__((ext_vector_type(8)))  __bf16 v8bf;
typedef __attribute__((ext_vector_type(8)))  float  v8f;

__device__ __forceinline__ v16bf cat16(v8bf lo, v8bf hi) {
  return __builtin_shufflevector(lo, hi, 0,1,2,3,4,5,6,7,8,9,10,11,12,13,14,15);
}

// monotonic float -> uint key (order-preserving, invertible)
__device__ __forceinline__ unsigned fkey(float f) {
  unsigned u = __float_as_uint(f);
  return (u & 0x80000000u) ? ~u : (u | 0x80000000u);
}

// ---------------------------------------------------------------------------
// K1: encoder GEMM. Block tile 128 rows x 128 latents, 8 waves.
// Wave w owns rows [16w,16w+16) and loops 8 n-subtiles of 16 latents.
// K staged in chunks of 32 (bf16 in LDS, rows padded to 40 halves = 80B so the
// 16 per-lane fragment rows land in distinct bank groups).
// ---------------------------------------------------------------------------
__global__ __launch_bounds__(256) void enc_gemm(
    const float* __restrict__ x, const float* __restrict__ Wenc,
    const float* __restrict__ benc, const float* __restrict__ bdec,
    float* __restrict__ abuf)
{
  __shared__ __bf16 ldsA[128 * 40];
  __shared__ __bf16 ldsB[128 * 40];

  const int t    = threadIdx.x;
  const int lane = t & 31;
  const int w    = t >> 5;
  const int row0 = blockIdx.y * 128;   // batch rows
  const int lat0 = blockIdx.x * 128;   // latent cols

  v8f acc[8] = {};

  #pragma unroll 1
  for (int kc = 0; kc < FEATS / 32; ++kc) {
    const int k0 = kc * 32;
    __syncthreads();   // previous iteration's LDS reads done

    // ---- stage A (xbar) and B (W_enc) chunks: fp32 global -> bf16 LDS ----
    #pragma unroll
    for (int i = 0; i < 4; ++i) {
      int q  = t + i * 256;        // float4 id in 128x32 chunk
      int rr = q >> 3;
      int c4 = (q & 7) * 4;
      float4 xv = *(const float4*)(x + (size_t)(row0 + rr) * FEATS + k0 + c4);
      float4 bd = *(const float4*)(bdec + k0 + c4);
      __bf16* da = &ldsA[rr * 40 + c4];
      da[0] = (__bf16)(xv.x - bd.x); da[1] = (__bf16)(xv.y - bd.y);
      da[2] = (__bf16)(xv.z - bd.z); da[3] = (__bf16)(xv.w - bd.w);
      float4 wv = *(const float4*)(Wenc + (size_t)(lat0 + rr) * FEATS + k0 + c4);
      __bf16* db = &ldsB[rr * 40 + c4];
      db[0] = (__bf16)wv.x; db[1] = (__bf16)wv.y;
      db[2] = (__bf16)wv.z; db[3] = (__bf16)wv.w;
    }
    __syncthreads();

    // ---- A fragment: lane m=lane&15, K runs [kb,kb+8) and [kb+16,kb+24) ----
    const int kbA = (lane & 16) ? 8 : 0;
    const __bf16* ar = &ldsA[(16 * w + (lane & 15)) * 40];
    v16bf afrag = cat16(*(const v8bf*)(ar + kbA), *(const v8bf*)(ar + kbA + 16));

    // ---- B fragments: lane n=lane&15, K runs [kb,kb+8) and [kb+16,kb+24) ----
    const int kbB = (lane & 16) ? 8 : 0;
    #pragma unroll
    for (int n = 0; n < 8; ++n) {
      const __bf16* br = &ldsB[(16 * n + (lane & 15)) * 40 + kbB];
      v16bf bfrag = cat16(*(const v8bf*)br, *(const v8bf*)(br + 16));
      acc[n] = __builtin_amdgcn_wmma_f32_16x16x32_bf16(
          false, afrag, false, bfrag, (short)0, acc[n], false, false);
    }
  }

  // ---- epilogue: +b_enc, store fp32 a-tile into f-buffer scratch ----
  const int mrow = row0 + 16 * w + ((lane & 16) ? 8 : 0);
  #pragma unroll
  for (int n = 0; n < 8; ++n) {
    const int col  = lat0 + 16 * n + (lane & 15);
    const float bb = benc[col];
    #pragma unroll
    for (int vr = 0; vr < 8; ++vr)
      abuf[(size_t)(mrow + vr) * LAT + col] = acc[n][vr] + bb;
  }
}

// ---------------------------------------------------------------------------
// K2: per-row top-32 via radix select on key's top 11 bits (2048 buckets).
// Row data stays in L2 between the 4 passes (64KB row). Deterministic:
// kmin is a value (tie-independent), ws compaction is index-ordered.
// ---------------------------------------------------------------------------
__global__ __launch_bounds__(256) void topk_kernel(
    float* __restrict__ abuf, float* __restrict__ ws_val, int* __restrict__ ws_idx)
{
  __shared__ unsigned hist[2048];
  __shared__ unsigned cand[1024];
  __shared__ unsigned cnts[256];
  __shared__ unsigned sc_tb, sc_need, sc_kmin, sc_candcnt, sc_total;

  const int t = threadIdx.x;
  const size_t row = blockIdx.x;
  float* __restrict__ arow = abuf + row * (size_t)LAT;

  for (int i = t; i < 2048; i += 256) hist[i] = 0;
  if (t == 0) sc_candcnt = 0;
  __syncthreads();

  // pass 1: histogram
  const float4* ag = (const float4*)arow;
  for (int i = t; i < LAT / 4; i += 256) {
    float4 v = ag[i];
    atomicAdd(&hist[fkey(v.x) >> 21], 1u);
    atomicAdd(&hist[fkey(v.y) >> 21], 1u);
    atomicAdd(&hist[fkey(v.z) >> 21], 1u);
    atomicAdd(&hist[fkey(v.w) >> 21], 1u);
  }
  __syncthreads();

  // threshold bucket (scan from top, early exit)
  if (t == 0) {
    unsigned cum = 0;
    for (int b = 2047; b >= 0; --b) {
      unsigned h = hist[b];
      if (cum + h >= TOPK) { sc_tb = (unsigned)b; sc_need = TOPK - cum; break; }
      cum += h;
    }
  }
  __syncthreads();
  const unsigned tb = sc_tb, need = sc_need;

  // pass 2: collect boundary-bucket keys
  for (int i = t; i < LAT; i += 256) {
    unsigned k = fkey(arow[i]);
    if ((k >> 21) == tb) {
      unsigned p = atomicAdd(&sc_candcnt, 1u);
      if (p < 1024) cand[p] = k;
    }
  }
  __syncthreads();

  // kmin = need-th largest key within boundary bucket (value is order-invariant)
  if (sc_candcnt > 1024) {
    if (t == 0) sc_kmin = tb << 21;   // pathological tie overflow: take bucket
  } else {
    const unsigned nc = sc_candcnt;
    for (unsigned e = t; e < nc; e += 256) {
      const unsigned ke = cand[e];
      unsigned rank = 0;
      for (unsigned j = 0; j < nc; ++j) {
        const unsigned kj = cand[j];
        rank += (kj > ke) || (kj == ke && j < e);
      }
      if (rank == need - 1) sc_kmin = ke;
    }
  }
  __syncthreads();
  const unsigned kmin = sc_kmin;

  // pass 3: deterministic index-ordered compaction of (idx, relu(val)) to ws
  const int base = t * (LAT / 256);
  unsigned c = 0;
  for (int i = 0; i < LAT / 256; ++i) c += (fkey(arow[base + i]) >= kmin);
  cnts[t] = c;
  __syncthreads();
  if (t == 0) {
    unsigned run = 0;
    for (int j = 0; j < 256; ++j) { unsigned tmp = cnts[j]; cnts[j] = run; run += tmp; }
    sc_total = run < TOPK ? run : TOPK;
  }
  __syncthreads();
  unsigned p = cnts[t];
  for (int i = 0; i < LAT / 256; ++i) {
    float v = arow[base + i];
    if (fkey(v) >= kmin) {
      if (p < TOPK) {
        ws_val[row * TOPK + p] = fmaxf(v, 0.f);
        ws_idx[row * TOPK + p] = base + i;
      }
      ++p;
    }
  }
  if (t < TOPK && (unsigned)t >= sc_total) {   // pad unused slots
    ws_val[row * TOPK + t] = 0.f;
    ws_idx[row * TOPK + t] = 0;
  }
  __syncthreads();   // all reads of original arow complete

  // pass 4: rewrite row in place: f = relu(val) where key >= kmin else 0
  float4* agw = (float4*)arow;
  for (int i = t; i < LAT / 4; i += 256) {
    float4 v = ag[i], o;
    o.x = (fkey(v.x) >= kmin) ? fmaxf(v.x, 0.f) : 0.f;
    o.y = (fkey(v.y) >= kmin) ? fmaxf(v.y, 0.f) : 0.f;
    o.z = (fkey(v.z) >= kmin) ? fmaxf(v.z, 0.f) : 0.f;
    o.w = (fkey(v.w) >= kmin) ? fmaxf(v.w, 0.f) : 0.f;
    agw[i] = o;
  }
}

// ---------------------------------------------------------------------------
// K0: W_dec [F,L] -> W_decT [L,F] (32x32 LDS tiles), enables coalesced decode.
// ---------------------------------------------------------------------------
__global__ __launch_bounds__(256) void transpose_wdec(
    const float* __restrict__ Wdec, float* __restrict__ WdecT)
{
  __shared__ float tile[32][33];
  const int tx = threadIdx.x & 31, ty = threadIdx.x >> 5;
  const int l0 = blockIdx.x * 32;
  const int f0 = blockIdx.y * 32;
  #pragma unroll
  for (int i = 0; i < 4; ++i)
    tile[ty + i * 8][tx] = Wdec[(size_t)(f0 + ty + i * 8) * LAT + l0 + tx];
  __syncthreads();
  #pragma unroll
  for (int i = 0; i < 4; ++i)
    WdecT[(size_t)(l0 + ty + i * 8) * FEATS + f0 + tx] = tile[tx][ty + i * 8];
}

// ---------------------------------------------------------------------------
// K3: sparse decode. One block per row; thread t covers features [4t,4t+4).
// ---------------------------------------------------------------------------
__global__ __launch_bounds__(256) void dec_kernel(
    const float* __restrict__ ws_val, const int* __restrict__ ws_idx,
    const float* __restrict__ Wdec, const float* __restrict__ WdecT,
    const float* __restrict__ bdec, float* __restrict__ xhat, int use_t)
{
  __shared__ float sval[TOPK];
  __shared__ int   sidx[TOPK];
  const int t = threadIdx.x;
  const size_t row = blockIdx.x;
  if (t < TOPK) {
    sval[t] = ws_val[row * TOPK + t];
    sidx[t] = ws_idx[row * TOPK + t];
  }
  __syncthreads();

  float4 acc = ((const float4*)bdec)[t];
  if (use_t) {
    #pragma unroll 4
    for (int j = 0; j < TOPK; ++j) {
      const float v = sval[j];
      if (v != 0.f) {
        float4 wv = *(const float4*)(WdecT + (size_t)sidx[j] * FEATS + t * 4);
        acc.x += v * wv.x; acc.y += v * wv.y; acc.z += v * wv.z; acc.w += v * wv.w;
      }
    }
  } else {
    for (int j = 0; j < TOPK; ++j) {
      const float v = sval[j];
      if (v != 0.f) {
        const float* wp = Wdec + (size_t)(t * 4) * LAT + sidx[j];
        acc.x += v * wp[0];
        acc.y += v * wp[(size_t)LAT];
        acc.z += v * wp[(size_t)2 * LAT];
        acc.w += v * wp[(size_t)3 * LAT];
      }
    }
  }
  ((float4*)(xhat + row * (size_t)FEATS))[t] = acc;
}

// ---------------------------------------------------------------------------
extern "C" void kernel_launch(void* const* d_in, const int* in_sizes, int n_in,
                              void* d_out, int out_size, void* d_ws, size_t ws_size,
                              hipStream_t stream)
{
  const float* x    = (const float*)d_in[0];
  const float* Wenc = (const float*)d_in[1];
  const float* benc = (const float*)d_in[2];
  const float* Wdec = (const float*)d_in[3];
  const float* bdec = (const float*)d_in[4];
  // d_in[5] = activate (k); fixed at 32 per the reference setup.

  float* fbuf = (float*)d_out;                       // [NBATCH, LAT]
  float* xhat = fbuf + (size_t)NBATCH * LAT;         // [NBATCH, FEATS]

  float* ws_val = (float*)d_ws;                      // NBATCH*TOPK
  int*   ws_idx = (int*)d_ws + (size_t)NBATCH * TOPK;
  const size_t wd_off = (size_t)NBATCH * TOPK * 8;   // 1 MB
  float* WdecT = (float*)((char*)d_ws + wd_off);
  const int use_t = (ws_size >= wd_off + (size_t)FEATS * LAT * 4) ? 1 : 0;

  if (use_t)
    transpose_wdec<<<dim3(LAT / 32, FEATS / 32), 256, 0, stream>>>(Wdec, WdecT);

  enc_gemm<<<dim3(LAT / 128, NBATCH / 128), 256, 0, stream>>>(x, Wenc, benc, bdec, fbuf);
  topk_kernel<<<NBATCH, 256, 0, stream>>>(fbuf, ws_val, ws_idx);
  dec_kernel<<<NBATCH, 256, 0, stream>>>(ws_val, ws_idx, Wdec,
                                         use_t ? WdecT : Wdec, bdec, xhat, use_t);
}
